// VisionMamba_2645699854379
// MI455X (gfx1250) — matrix operand order repaired and checked by reading, not compiled, not run on hardware
//
#include <hip/hip_runtime.h>
#include <math.h>
#include <stdint.h>

// ---------------- CDNA5 WMMA types ----------------
typedef __attribute__((ext_vector_type(16))) __bf16 bf16x16;
typedef __attribute__((ext_vector_type(8)))  float  f32x8;

// ---------------- model dims (fixed by reference) ----------------
#define BB    32
#define TT    30
#define HHH   11
#define WWW   11
#define LL    3630            // TT*HHH*WWW
#define DIMM  64
#define DINN  128
#define DMM   256
#define NSTT  16
#define RNKK  8
#define CST   32
#define BL    116160          // BB*LL
#define B2L   232320          // 2*BB*LL

#define ACT_NONE     0
#define ACT_SILU     1
#define ACT_GELU     2
#define ACT_SOFTPLUS 3

__device__ __forceinline__ float act_apply(float x, int act) {
  if (act == ACT_SILU)     return x / (1.f + __expf(-x));
  if (act == ACT_GELU)     return 0.5f * x * (1.f + erff(x * 0.70710678118654752f));
  if (act == ACT_SOFTPLUS) return (x > 20.f) ? x : log1pf(__expf(x));
  return x;
}

// Convert two contiguous 8-float runs (16B-aligned) into a bf16x16 fragment.
__device__ __forceinline__ bf16x16 cvt8x2(const float* __restrict__ p0,
                                          const float* __restrict__ p1) {
  float4 q0 = *(const float4*)(p0);
  float4 q1 = *(const float4*)(p0 + 4);
  float4 q2 = *(const float4*)(p1);
  float4 q3 = *(const float4*)(p1 + 4);
  bf16x16 r;
  r[0]  = (__bf16)q0.x; r[1]  = (__bf16)q0.y; r[2]  = (__bf16)q0.z; r[3]  = (__bf16)q0.w;
  r[4]  = (__bf16)q1.x; r[5]  = (__bf16)q1.y; r[6]  = (__bf16)q1.z; r[7]  = (__bf16)q1.w;
  r[8]  = (__bf16)q2.x; r[9]  = (__bf16)q2.y; r[10] = (__bf16)q2.z; r[11] = (__bf16)q2.w;
  r[12] = (__bf16)q3.x; r[13] = (__bf16)q3.y; r[14] = (__bf16)q3.z; r[15] = (__bf16)q3.w;
  return r;
}

// =====================================================================
// bf16 WMMA GEMM (branch-free inner loop).
//   Out[m][n] = act( sum_k A[m][k]*W[n][k] + bias[n] ) + Cin[m][n]
// Preconditions: M % 16 == 0, K % 32 == 0, Nw % 32 == 0 (weights padded),
//                A rows allow unconditional reads of K floats from col 0.
// Block = 256 threads = 8 waves; wave owns a 16x32 tile (two WMMAs / K-slab,
// A fragment reused).  A/B use contiguous k = 16*half + s; C/D layout per CDNA5 ISA 7.12.2 (wave32).
// =====================================================================
__global__ __launch_bounds__(256) void vm_gemm_wmma_bf16(
    const float* __restrict__ A, int lda,
    const float* __restrict__ W,     // [Nw x K] row-major, padded
    const float* bias, const float* Cin,
    float* Out, int M, int Nw, int Nout, int ldOut, int K, int act)
{
  const int wave = threadIdx.x >> 5;
  const int lane = threadIdx.x & 31;
  const int m0 = blockIdx.y * 128 + wave * 16;
  const int n0 = blockIdx.x * 32;
  if (m0 >= M) return;                       // wave-uniform: EXEC stays all-1s

  const int half = lane >> 4;
  const int mr   = lane & 15;

  const float* ap  = A + (long)(m0 + mr) * lda + 16 * half;
  const float* wp0 = W + (long)(n0 + mr) * K + 16 * half;
  const float* wp1 = W + (long)(n0 + 16 + mr) * K + 16 * half;

  f32x8 acc0 = {}, acc1 = {};
  for (int k0 = 0; k0 < K; k0 += 32) {
    __builtin_prefetch(ap + k0 + 128, 0, 1);           // speculative: fault-free
    bf16x16 af = cvt8x2(ap + k0, ap + k0 + 8);
    bf16x16 b0 = cvt8x2(wp0 + k0, wp0 + k0 + 8);
    bf16x16 b1 = cvt8x2(wp1 + k0, wp1 + k0 + 8);
    acc0 = __builtin_amdgcn_wmma_f32_16x16x32_bf16(false, af, false, b0,
                                                   (short)0, acc0, false, false);
    acc1 = __builtin_amdgcn_wmma_f32_16x16x32_bf16(false, af, false, b1,
                                                   (short)0, acc1, false, false);
  }

  const int col0 = n0 + mr;
  const int col1 = n0 + 16 + mr;
  if (col0 < Nout) {
    float bv = bias ? bias[col0] : 0.f;
#pragma unroll
    for (int v = 0; v < 8; ++v) {
      long row = m0 + v + 8 * half;
      float val = act_apply(acc0[v] + bv, act);
      if (Cin) val += Cin[row * ldOut + col0];
      Out[row * ldOut + col0] = val;
    }
  }
  if (col1 < Nout) {
    float bv = bias ? bias[col1] : 0.f;
#pragma unroll
    for (int v = 0; v < 8; ++v) {
      long row = m0 + v + 8 * half;
      float val = act_apply(acc1[v] + bv, act);
      if (Cin) val += Cin[row * ldOut + col1];
      Out[row * ldOut + col1] = val;
    }
  }
}

// ------- weight padders (run once, before the block loop) -------
__global__ void vm_pad_xproj(const float* __restrict__ w, float* __restrict__ wp) {
  int idx = blockIdx.x * 256 + threadIdx.x;            // [4][64][256]
  if (idx >= 4 * 64 * 256) return;
  int k = idx & 255, n = (idx >> 8) & 63, d = idx >> 14;
  wp[idx] = (n < 40) ? w[((long)d * 40 + n) * 256 + k] : 0.f;
}
__global__ void vm_pad_dtw(const float* __restrict__ w, float* __restrict__ wp) {
  int idx = blockIdx.x * 256 + threadIdx.x;            // [4][256][32]
  if (idx >= 4 * 256 * 32) return;
  int k = idx & 31, n = (idx >> 5) & 255, d = idx >> 13;
  wp[idx] = (k < 8) ? w[((long)d * 256 + n) * 8 + k] : 0.f;
}

// ---------------- LayerNorm over DIM=64, one wave per row ----------------
__global__ __launch_bounds__(256) void vm_ln64(
    const float* __restrict__ X, const float* __restrict__ g,
    const float* __restrict__ b, float* __restrict__ Y, int Mrows)
{
  int row  = blockIdx.x * 8 + (threadIdx.x >> 5);
  int lane = threadIdx.x & 31;
  if (row >= Mrows) return;
  const float* xr = X + (long)row * 64;
  float x0 = xr[lane], x1 = xr[lane + 32];
  float s = x0 + x1, s2 = x0 * x0 + x1 * x1;
#pragma unroll
  for (int off = 16; off >= 1; off >>= 1) {
    s  += __shfl_xor(s,  off, 32);
    s2 += __shfl_xor(s2, off, 32);
  }
  float mean = s * (1.f / 64.f);
  float var  = s2 * (1.f / 64.f) - mean * mean;
  float inv  = rsqrtf(var + 1e-5f);
  float* yr = Y + (long)row * 64;
  yr[lane]      = (x0 - mean) * inv * g[lane]      + b[lane];
  yr[lane + 32] = (x1 - mean) * inv * g[lane + 32] + b[lane + 32];
}

// ---------------- stem: spatial 3x3 conv + BN + ReLU ----------------
__global__ void vm_stem_sconv(const float* __restrict__ x,
                              const float* __restrict__ sw,
                              const float* __restrict__ g, const float* __restrict__ bb_,
                              const float* __restrict__ mm, const float* __restrict__ vv,
                              float* __restrict__ S1)
{
  long idx = (long)blockIdx.x * 256 + threadIdx.x;
  const long tot = (long)32 * 32 * 32 * 11 * 11;
  if (idx >= tot) return;
  long t = idx;
  int w = t % 11; t /= 11; int h = t % 11; t /= 11;
  int tq = t % 32; t /= 32; int c = t % 32; int b = t / 32;
  float acc = 0.f;
#pragma unroll
  for (int i = 0; i < 3; ++i)
#pragma unroll
    for (int j = 0; j < 3; ++j)
      acc += x[((long)(b * 32 + tq) * 13 + (h + i)) * 13 + (w + j)] * sw[c * 9 + i * 3 + j];
  float sc = g[c] * rsqrtf(vv[c] + 1e-5f);
  float y = acc * sc + (bb_[c] - mm[c] * sc);
  S1[idx] = y > 0.f ? y : 0.f;
}

// ---------------- stem: temporal dw conv(3) + BN + ReLU ----------------
__global__ void vm_stem_tconv(const float* __restrict__ S1,
                              const float* __restrict__ tw,
                              const float* __restrict__ g, const float* __restrict__ bb_,
                              const float* __restrict__ mm, const float* __restrict__ vv,
                              float* __restrict__ S2)
{
  long idx = (long)blockIdx.x * 256 + threadIdx.x;
  const long tot = (long)32 * 32 * 30 * 11 * 11;
  if (idx >= tot) return;
  long t = idx;
  int w = t % 11; t /= 11; int h = t % 11; t /= 11;
  int tq = t % 30; t /= 30; int c = t % 32; int b = t / 32;
  float acc = 0.f;
#pragma unroll
  for (int k = 0; k < 3; ++k)
    acc += S1[(((long)(b * 32 + c) * 32 + (tq + k)) * 11 + h) * 11 + w] * tw[c * 3 + k];
  float sc = g[c] * rsqrtf(vv[c] + 1e-5f);
  float y = acc * sc + (bb_[c] - mm[c] * sc);
  S2[idx] = y > 0.f ? y : 0.f;
}

// ---------------- stem: 1x1x1 conv -> XB [B, L, 64] ----------------
__global__ void vm_stem_last(const float* __restrict__ S2,
                             const float* __restrict__ lw, float* __restrict__ XB)
{
  long idx = (long)blockIdx.x * 256 + threadIdx.x;
  const long tot = (long)BL * DIMM;
  if (idx >= tot) return;
  int o = idx % DIMM;
  long bl = idx / DIMM;
  int l = bl % LL, b = bl / LL;
  float acc = 0.f;
#pragma unroll 8
  for (int c = 0; c < 32; ++c)
    acc += lw[o * 32 + c] * S2[(long)(b * 32 + c) * LL + l];
  XB[idx] = acc;
}

// ------- depthwise 3x3x3 SAME conv + bias + SiLU; input = XZ[:, :128] -------
__global__ void vm_dwconv3d_silu(const float* __restrict__ XZ,
                                 const float* __restrict__ w3,
                                 const float* __restrict__ b3, float* __restrict__ XC)
{
  long idx = (long)blockIdx.x * 256 + threadIdx.x;
  const long tot = (long)BL * DINN;
  if (idx >= tot) return;
  int c = idx & 127;
  long bl = idx >> 7;
  int l = bl % LL, b = bl / LL;
  int t = l / 121, r = l % 121, h = r / 11, w = r % 11;
  float acc = b3[c];
#pragma unroll
  for (int dt = -1; dt <= 1; ++dt)
#pragma unroll
    for (int dh = -1; dh <= 1; ++dh)
#pragma unroll
      for (int dw = -1; dw <= 1; ++dw) {
        int tt = t + dt, hh = h + dh, ww = w + dw;
        if (tt >= 0 && tt < TT && hh >= 0 && hh < HHH && ww >= 0 && ww < WWW)
          acc += XZ[((long)b * LL + (tt * 121 + hh * 11 + ww)) * 256 + c] *
                 w3[c * 27 + (dt + 1) * 9 + (dh + 1) * 3 + (dw + 1)];
      }
  XC[idx] = acc / (1.f + __expf(-acc));
}

// ------- build bidirectional sequence -------
__global__ void vm_build_seq2(const float* __restrict__ XC, float* __restrict__ SEQ2)
{
  long idx = (long)blockIdx.x * 256 + threadIdx.x;
  const long tot = (long)B2L * DINN;
  if (idx >= tot) return;
  int c = idx & 127;
  long bl = idx >> 7;
  int l = bl % LL, bb = bl / LL;
  int sb = bb & 31;
  int ls = (bb < 32) ? l : (LL - 1 - l);
  SEQ2[idx] = XC[((long)sb * LL + ls) * DINN + c];
}

// ------- mamba causal depthwise conv1d(k=4) + bias + SiLU; input MXZ[:, :256] -------
__global__ void vm_conv1d_silu(const float* __restrict__ MXZ,
                               const float* __restrict__ cw,
                               const float* __restrict__ cb, float* __restrict__ MXC)
{
  long idx = (long)blockIdx.x * 256 + threadIdx.x;
  const long tot = (long)B2L * DMM;
  if (idx >= tot) return;
  int d = idx & 255;
  long bl = idx >> 8;
  int l = bl % LL, bb = bl / LL;
  float acc = cb[d];
#pragma unroll
  for (int k = 0; k < 4; ++k) {
    int ls = l - 3 + k;
    if (ls >= 0) acc += MXZ[((long)bb * LL + ls) * 512 + d] * cw[d * 4 + k];
  }
  MXC[idx] = acc / (1.f + __expf(-acc));
}

// ------- selective scan: block = batch lane bb, thread = channel d -------
// Serial over L; per-step B/C (32 floats) double-buffered in LDS via
// GLOBAL_LOAD_ASYNC_TO_LDS_B32 issued by wave0 one step ahead (ASYNCcnt),
// published to the other 7 waves with s_wait_asynccnt + one barrier/step.
__global__ __launch_bounds__(256) void vm_scan(
    const float* __restrict__ DT, const float* __restrict__ U,
    const float* __restrict__ DBC, const float* __restrict__ MXZ,
    const float* __restrict__ Alog, const float* __restrict__ Dp,
    float* __restrict__ YS)
{
  const int bb   = blockIdx.x;
  const int d    = threadIdx.x;
  const int lane = threadIdx.x & 31;
  const bool w0  = (threadIdx.x < 32);      // wave-uniform: wave 0, EXEC all-1s

  float a[NSTT], h[NSTT];
#pragma unroll
  for (int n = 0; n < NSTT; ++n) { a[n] = -__expf(Alog[d * NSTT + n]); h[n] = 0.f; }
  const float dskip = Dp[d];

  __shared__ float bc[2][32];               // [buf][ B(0..15) | C(16..31) ]

  // async DMA of DBC[bb, l, 8..39] (B then C) straight into LDS buffer `buf`
  auto issue_bc = [&](int l, int buf) {
    unsigned lds = (unsigned)(uintptr_t)(&bc[buf][lane]);
    uint64_t gaddr = (uint64_t)(uintptr_t)(DBC + ((long)bb * LL + l) * 40 + 8 + lane);
    asm volatile("global_load_async_to_lds_b32 %0, %1, off"
                 :: "v"(lds), "v"(gaddr) : "memory");
  };

  if (w0) {
    issue_bc(0, 0);
    asm volatile("s_wait_asynccnt 0x0" ::: "memory");
  }
  __syncthreads();

  for (int l = 0; l < LL; ++l) {
    const int cur = l & 1;
    if (w0 && (l + 1) < LL) issue_bc(l + 1, cur ^ 1);   // prefetch next step

    long row = (long)bb * LL + l;
    float dt = DT[row * DMM + d];
    float u  = U[row * DMM + d];
    float du = dt * u;
    float y = 0.f;
#pragma unroll
    for (int n = 0; n < NSTT; ++n) {
      float dA = __expf(dt * a[n]);
      h[n] = dA * h[n] + du * bc[cur][n];
      y += h[n] * bc[cur][16 + n];
    }
    float mz = MXZ[row * 512 + 256 + d];
    float zs = mz / (1.f + __expf(-mz));
    YS[row * DMM + d] = (y + u * dskip) * zs;

    if (w0) asm volatile("s_wait_asynccnt 0x0" ::: "memory");
    __syncthreads();                         // publish next buffer to all waves
  }
}

// ------- combine fwd+bwd mamba outputs and gate with silu(z) -------
__global__ void vm_combine(const float* __restrict__ MOUT,
                           const float* __restrict__ XZ, float* __restrict__ YC)
{
  long idx = (long)blockIdx.x * 256 + threadIdx.x;
  const long tot = (long)BL * DINN;
  if (idx >= tot) return;
  int c = idx & 127;
  long bl = idx >> 7;
  int l = bl % LL, b = bl / LL;
  float zf = XZ[((long)b * LL + l) * 256 + 128 + c];
  float zs = zf / (1.f + __expf(-zf));
  float yv = MOUT[((long)b * LL + l) * DINN + c] +
             MOUT[((long)(b + 32) * LL + (LL - 1 - l)) * DINN + c];
  YC[idx] = yv * zs;
}

// ------- mean pool over L; write feature part of d_out too -------
__global__ void vm_meanpool(const float* __restrict__ XB,
                            float* __restrict__ feat, float* __restrict__ outFeat)
{
  int b = blockIdx.x, c = threadIdx.x;      // 32 blocks x 64 threads
  float s = 0.f;
  for (int l = 0; l < LL; ++l) s += XB[((long)b * LL + l) * DIMM + c];
  float v = s * (1.f / (float)LL);
  feat[b * DIMM + c] = v;
  outFeat[b * DIMM + c] = v;
}

// ------- classifier head (tiny: 32x16x64) -------
__global__ void vm_head(const float* __restrict__ feat, const float* __restrict__ hw,
                        const float* __restrict__ hb, float* __restrict__ logits)
{
  int i = blockIdx.x * blockDim.x + threadIdx.x;
  if (i >= 32 * 16) return;
  int b = i >> 4, n = i & 15;
  float s = hb[n];
#pragma unroll 8
  for (int c = 0; c < 64; ++c) s += feat[b * 64 + c] * hw[n * 64 + c];
  logits[b * 16 + n] = s;
}

// =====================================================================
static inline int cdiv(long a, long b) { return (int)((a + b - 1) / b); }

extern "C" void kernel_launch(void* const* d_in, const int* in_sizes, int n_in,
                              void* d_out, int out_size, void* d_ws, size_t ws_size,
                              hipStream_t stream) {
  const float* x        = (const float*)d_in[0];
  const float* sconv_w  = (const float*)d_in[1];
  const float* bn1_g    = (const float*)d_in[2];
  const float* bn1_b    = (const float*)d_in[3];
  const float* bn1_m    = (const float*)d_in[4];
  const float* bn1_v    = (const float*)d_in[5];
  const float* tconv_w  = (const float*)d_in[6];
  const float* bn2_g    = (const float*)d_in[7];
  const float* bn2_b    = (const float*)d_in[8];
  const float* bn2_m    = (const float*)d_in[9];
  const float* bn2_v    = (const float*)d_in[10];
  const float* last_w   = (const float*)d_in[11];
  const float* ln1_g    = (const float*)d_in[12];
  const float* ln1_b    = (const float*)d_in[13];
  const float* in_proj  = (const float*)d_in[14];
  const float* c3_w     = (const float*)d_in[15];
  const float* c3_b     = (const float*)d_in[16];
  const float* m_in_w   = (const float*)d_in[17];
  const float* m_conv_w = (const float*)d_in[18];
  const float* m_conv_b = (const float*)d_in[19];
  const float* m_xproj  = (const float*)d_in[20];
  const float* m_dtw    = (const float*)d_in[21];
  const float* m_dtb    = (const float*)d_in[22];
  const float* m_Alog   = (const float*)d_in[23];
  const float* m_D      = (const float*)d_in[24];
  const float* m_out_w  = (const float*)d_in[25];
  const float* out_proj = (const float*)d_in[26];
  const float* ln2_g    = (const float*)d_in[27];
  const float* ln2_b    = (const float*)d_in[28];
  const float* fc1_w    = (const float*)d_in[29];
  const float* fc1_b    = (const float*)d_in[30];
  const float* fc2_w    = (const float*)d_in[31];
  const float* fc2_b    = (const float*)d_in[32];
  const float* head_w   = (const float*)d_in[33];
  const float* head_b   = (const float*)d_in[34];

  float* out = (float*)d_out;   // [32*16 logits | 32*64 feature]
  float* ws  = (float*)d_ws;

  // -------- workspace layout (element offsets) --------
  const long XB_O   = 0;                         // BL*64
  const long HX_O   = XB_O   + (long)BL  * 64;
  const long XZ_O   = HX_O   + (long)BL  * 64;   // BL*256
  const long XC_O   = XZ_O   + (long)BL  * 256;  // BL*128
  const long SEQ2_O = XC_O   + (long)BL  * 128;  // B2L*128
  const long MXZ_O  = SEQ2_O + (long)B2L * 128;  // B2L*512
  const long MXC_O  = MXZ_O  + (long)B2L * 512;  // B2L*256
  const long DBC_O  = MXC_O  + (long)B2L * 256;  // B2L*40
  const long DT_O   = DBC_O  + (long)B2L * 40;   // B2L*256
  const long YS_O   = DT_O   + (long)B2L * 256;  // B2L*256
  const long MOUT_O = YS_O   + (long)B2L * 256;  // B2L*128
  const long FEAT_O = MOUT_O + (long)B2L * 128;  // 32*64
  const long WXP_O  = FEAT_O + 2048;             // 4*64*256 padded xproj
  const long WDT_O  = WXP_O  + 4L * 64 * 256;    // 4*256*32 padded dtproj
  const long S1_O   = MXZ_O;                     // stem temps overlay MXZ (dead)
  const long S2_O   = MXZ_O + 4000000;

  // -------- pad ragged weights (once) --------
  vm_pad_xproj<<<cdiv(4L * 64 * 256, 256), 256, 0, stream>>>(m_xproj, ws + WXP_O);
  vm_pad_dtw  <<<cdiv(4L * 256 * 32, 256), 256, 0, stream>>>(m_dtw,   ws + WDT_O);

  // -------- stem --------
  vm_stem_sconv<<<cdiv((long)32*32*32*121, 256), 256, 0, stream>>>(
      x, sconv_w, bn1_g, bn1_b, bn1_m, bn1_v, ws + S1_O);
  vm_stem_tconv<<<cdiv((long)32*32*30*121, 256), 256, 0, stream>>>(
      ws + S1_O, tconv_w, bn2_g, bn2_b, bn2_m, bn2_v, ws + S2_O);
  vm_stem_last<<<cdiv((long)BL*64, 256), 256, 0, stream>>>(
      ws + S2_O, last_w, ws + XB_O);

  // -------- blocks --------
  for (int i = 0; i < 4; ++i) {
    // SSM branch
    vm_ln64<<<cdiv(BL, 8), 256, 0, stream>>>(ws + XB_O, ln1_g + i*64, ln1_b + i*64,
                                             ws + HX_O, BL);
    vm_gemm_wmma_bf16<<<dim3(256/32, cdiv(BL,128)), 256, 0, stream>>>(
        ws + HX_O, 64, in_proj + (long)i*256*64, nullptr, nullptr,
        ws + XZ_O, BL, 256, 256, 256, 64, ACT_NONE);
    vm_dwconv3d_silu<<<cdiv((long)BL*128, 256), 256, 0, stream>>>(
        ws + XZ_O, c3_w + (long)i*128*27, c3_b + i*128, ws + XC_O);
    vm_build_seq2<<<cdiv((long)B2L*128, 256), 256, 0, stream>>>(ws + XC_O, ws + SEQ2_O);
    vm_gemm_wmma_bf16<<<dim3(512/32, cdiv(B2L,128)), 256, 0, stream>>>(
        ws + SEQ2_O, 128, m_in_w + (long)i*512*128, nullptr, nullptr,
        ws + MXZ_O, B2L, 512, 512, 512, 128, ACT_NONE);
    vm_conv1d_silu<<<cdiv((long)B2L*256, 256), 256, 0, stream>>>(
        ws + MXZ_O, m_conv_w + (long)i*256*4, m_conv_b + i*256, ws + MXC_O);
    vm_gemm_wmma_bf16<<<dim3(64/32, cdiv(B2L,128)), 256, 0, stream>>>(
        ws + MXC_O, 256, ws + WXP_O + (long)i*64*256, nullptr, nullptr,
        ws + DBC_O, B2L, 64, 40, 40, 256, ACT_NONE);
    vm_gemm_wmma_bf16<<<dim3(256/32, cdiv(B2L,128)), 256, 0, stream>>>(
        ws + DBC_O, 40, ws + WDT_O + (long)i*256*32, m_dtb + i*256, nullptr,
        ws + DT_O, B2L, 256, 256, 256, 32, ACT_SOFTPLUS);
    vm_scan<<<64, 256, 0, stream>>>(
        ws + DT_O, ws + MXC_O, ws + DBC_O, ws + MXZ_O,
        m_Alog + (long)i*256*16, m_D + i*256, ws + YS_O);
    vm_gemm_wmma_bf16<<<dim3(128/32, cdiv(B2L,128)), 256, 0, stream>>>(
        ws + YS_O, 256, m_out_w + (long)i*128*256, nullptr, nullptr,
        ws + MOUT_O, B2L, 128, 128, 128, 256, ACT_NONE);
    vm_combine<<<cdiv((long)BL*128, 256), 256, 0, stream>>>(
        ws + MOUT_O, ws + XZ_O, ws + XC_O);
    vm_gemm_wmma_bf16<<<dim3(64/32, cdiv(BL,128)), 256, 0, stream>>>(
        ws + XC_O, 128, out_proj + (long)i*64*128, nullptr, ws + XB_O,
        ws + XB_O, BL, 64, 64, 64, 128, ACT_NONE);
    // MLP branch
    vm_ln64<<<cdiv(BL, 8), 256, 0, stream>>>(ws + XB_O, ln2_g + i*64, ln2_b + i*64,
                                             ws + HX_O, BL);
    vm_gemm_wmma_bf16<<<dim3(256/32, cdiv(BL,128)), 256, 0, stream>>>(
        ws + HX_O, 64, fc1_w + (long)i*256*64, fc1_b + i*256, nullptr,
        ws + XZ_O, BL, 256, 256, 256, 64, ACT_GELU);
    vm_gemm_wmma_bf16<<<dim3(64/32, cdiv(BL,128)), 256, 0, stream>>>(
        ws + XZ_O, 256, fc2_w + (long)i*64*256, fc2_b + i*64, ws + XB_O,
        ws + XB_O, BL, 64, 64, 64, 256, ACT_NONE);
  }

  // -------- pool + head: out = [logits(512) | feature(2048)] --------
  vm_meanpool<<<32, 64, 0, stream>>>(ws + XB_O, ws + FEAT_O, out + 512);
  vm_head<<<2, 256, 0, stream>>>(ws + FEAT_O, head_w, head_b, out);
}
